// SeerAttnQwen3Attention_12094627906196
// MI455X (gfx1250) — matrix operand order repaired and checked by reading, not compiled, not run on hardware
//
#include <hip/hip_runtime.h>
#include <hip/hip_bf16.h>
#include <math.h>

#define T_    2048
#define HID_  2048
#define H_    16
#define KV_   8
#define G_    2
#define D_    128
#define GH_   128
#define BS_   64
#define NB_   32
#define START_ 409
#define SCALE_ 0.08838834764831845f   // 1/sqrt(128)

typedef __attribute__((ext_vector_type(16))) __bf16 v16bf;
typedef __attribute__((ext_vector_type(8)))  float  v8f;

// ---------------- elementwise helpers ----------------

__global__ void k_cvt_bf16(const float* __restrict__ x, __bf16* __restrict__ y, int n) {
  int i = blockIdx.x * 256 + threadIdx.x;
  if (i < n) y[i] = (__bf16)x[i];
}

// W[K][N] (row-major) -> Wt[N][K] bf16
__global__ void k_transpose_bf16(const float* __restrict__ W, __bf16* __restrict__ Wt, int K, int N) {
  int i = blockIdx.x * 256 + threadIdx.x;
  if (i < K * N) {
    int k = i / N, n = i % N;
    Wt[(size_t)n * K + k] = (__bf16)W[i];
  }
}

// ---------------- bf16 WMMA GEMM: C[M][N] = A[M][K] * Bt[N][K]^T ----------------
// block = 128 (4 waves in 2x2 quadrants of a 64x64 block tile);
// each wave computes a 32x32 tile via 4 accumulators (A/B fragment reuse).
// grid = (N/64, M/64); M, N multiples of 64.
__global__ __launch_bounds__(128) void k_gemm_bf16(const __bf16* __restrict__ A,
                                                   const __bf16* __restrict__ Bt,
                                                   float* __restrict__ C,
                                                   int M, int N, int K) {
  const int wave = threadIdx.x >> 5;
  const int lane = threadIdx.x & 31;
  const int m0 = blockIdx.y * 64 + (wave >> 1) * 32;
  const int n0 = blockIdx.x * 64 + (wave & 1) * 32;
  if (m0 >= M || n0 >= N) return;   // wave-uniform
  const int mrow = lane & 15;
  const int half = lane >> 4;
  v8f acc00 = {}, acc01 = {}, acc10 = {}, acc11 = {};
  for (int k0 = 0; k0 < K; k0 += 32) {
    v16bf a0, a1, b0, b1;
    // A 16x32 layout: lane half 0 -> K 0..15; half 1 -> K 16..31
    const __bf16* a0row = A + (size_t)(m0 + mrow) * K + k0;
    const __bf16* a1row = A + (size_t)(m0 + 16 + mrow) * K + k0;
#pragma unroll
    for (int j = 0; j < 8; ++j) {
      a0[j]     = a0row[half * 16 + j];
      a0[8 + j] = a0row[half * 16 + 8 + j];
      a1[j]     = a1row[half * 16 + j];
      a1[8 + j] = a1row[half * 16 + 8 + j];
    }
    // B 32x16 layout: lanes 0-15 hold K 0..15, lanes 16-31 hold K 16..31, column = lane%16
    const __bf16* b0row = Bt + (size_t)(n0 + mrow) * K + k0 + half * 16;
    const __bf16* b1row = Bt + (size_t)(n0 + 16 + mrow) * K + k0 + half * 16;
#pragma unroll
    for (int j = 0; j < 16; ++j) { b0[j] = b0row[j]; b1[j] = b1row[j]; }
    acc00 = __builtin_amdgcn_wmma_f32_16x16x32_bf16(false, a0, false, b0, (short)0, acc00, false, false);
    acc01 = __builtin_amdgcn_wmma_f32_16x16x32_bf16(false, a0, false, b1, (short)0, acc01, false, false);
    acc10 = __builtin_amdgcn_wmma_f32_16x16x32_bf16(false, a1, false, b0, (short)0, acc10, false, false);
    acc11 = __builtin_amdgcn_wmma_f32_16x16x32_bf16(false, a1, false, b1, (short)0, acc11, false, false);
  }
  // C/D layout: row = i + 8*half, col = lane%16
#pragma unroll
  for (int i = 0; i < 8; ++i) {
    C[(size_t)(m0 + i + 8 * half) * N + n0 + mrow]           = acc00[i];
    C[(size_t)(m0 + i + 8 * half) * N + n0 + 16 + mrow]      = acc01[i];
    C[(size_t)(m0 + 16 + i + 8 * half) * N + n0 + mrow]      = acc10[i];
    C[(size_t)(m0 + 16 + i + 8 * half) * N + n0 + 16 + mrow] = acc11[i];
  }
}

// ---------------- RMSNorm in place: rows x D_, one wave per row ----------------
__global__ void k_rmsnorm(float* __restrict__ x, const float* __restrict__ w, int rows) {
  int row = blockIdx.x;
  int lane = threadIdx.x;
  float* p = x + (size_t)row * D_;
  float v[4];
  float s = 0.f;
#pragma unroll
  for (int j = 0; j < 4; ++j) { v[j] = p[lane * 4 + j]; s += v[j] * v[j]; }
  for (int m = 1; m < 32; m <<= 1) s += __shfl_xor(s, m);
  float inv = rsqrtf(s * (1.0f / D_) + 1e-6f);
#pragma unroll
  for (int j = 0; j < 4; ++j) p[lane * 4 + j] = w[lane * 4 + j] * (v[j] * inv);
}

// q_pool[t][kv][d] = mean over g of q[t][kv*G+g][d]
__global__ void k_qpool(const float* __restrict__ q, float* __restrict__ qp) {
  int i = blockIdx.x * 256 + threadIdx.x;
  if (i >= T_ * KV_ * D_) return;
  int d = i % D_; int kv = (i / D_) % KV_; int t = i / (D_ * KV_);
  float s = 0.f;
#pragma unroll
  for (int g = 0; g < G_; ++g) s += q[((size_t)t * H_ + kv * G_ + g) * D_ + d];
  qp[i] = s * (1.0f / G_);
}

// k_blk[n][kv][d] = mean over 64 rows
__global__ void k_kblk(const float* __restrict__ k, float* __restrict__ kb) {
  int i = blockIdx.x * 256 + threadIdx.x;
  if (i >= NB_ * KV_ * D_) return;
  int d = i % D_; int kv = (i / D_) % KV_; int n = i / (D_ * KV_);
  float s = 0.f;
  for (int b = 0; b < BS_; ++b) s += k[((size_t)(n * BS_ + b) * KV_ + kv) * D_ + d];
  kb[i] = s * (1.0f / BS_);
}

// small fp32 gemm: C[rows][GH] = A[rows][D] * W[D][GH]
__global__ void k_small_gemm(const float* __restrict__ A, const float* __restrict__ W,
                             float* __restrict__ C, int rows) {
  int i = blockIdx.x * 256 + threadIdx.x;
  if (i >= rows * GH_) return;
  int g = i % GH_; int r = i / GH_;
  float s = 0.f;
  for (int d = 0; d < D_; ++d) s += A[(size_t)r * D_ + d] * W[(size_t)d * GH_ + g];
  C[i] = s;
}

// rope over GH-wide rows; pos = row / KV
__global__ void k_rope_gate(const float* __restrict__ raw, const float* __restrict__ cosT,
                            const float* __restrict__ sinT, float* __restrict__ out, int rows) {
  int i = blockIdx.x * 256 + threadIdx.x;
  if (i >= rows * GH_) return;
  int g = i % GH_; int r = i / GH_; int pos = r / KV_;
  float x = raw[i];
  float xr = (g < GH_ / 2) ? -raw[(size_t)r * GH_ + g + GH_ / 2]
                           :  raw[(size_t)r * GH_ + g - GH_ / 2];
  out[i] = x * cosT[(size_t)pos * GH_ + g] + xr * sinT[(size_t)pos * GH_ + g];
}

// predict_mask[kv][t][n]
__global__ void k_predict(const float* __restrict__ gq, const float* __restrict__ gk,
                          float* __restrict__ pm) {
  int i = blockIdx.x * 256 + threadIdx.x;
  if (i >= KV_ * T_ * NB_) return;
  int n = i % NB_; int t = (i / NB_) % T_; int kv = i / (NB_ * T_);
  float s = 0.f;
  for (int g = 0; g < GH_; ++g)
    s += gq[((size_t)t * KV_ + kv) * GH_ + g] * gk[((size_t)n * KV_ + kv) * GH_ + g];
  s *= 0.08838834764831845f;   // 1/sqrt(GH)
  pm[i] = (t / BS_ >= n) ? s : -1e9f;
}

// rope over D-wide rows (q or k), fp32 in, bf16 out; x[t][heads][D]
__global__ void k_rope_bf16(const float* __restrict__ x, const float* __restrict__ cosT,
                            const float* __restrict__ sinT, __bf16* __restrict__ out, int heads) {
  int i = blockIdx.x * 256 + threadIdx.x;
  int total = T_ * heads * D_;
  if (i >= total) return;
  int d = i % D_; int t = i / (D_ * heads);
  size_t base = (size_t)(i - d);
  float v = x[i];
  float vr = (d < 64) ? -x[base + d + 64] : x[base + d - 64];
  out[i] = (__bf16)(v * cosT[(size_t)t * D_ + d] + vr * sinT[(size_t)t * D_ + d]);
}

// vt[kv][d][t] = (bf16) v[t][kv][d]
__global__ void k_vt(const float* __restrict__ v, __bf16* __restrict__ vt) {
  int i = blockIdx.x * 256 + threadIdx.x;
  if (i >= T_ * KV_ * D_) return;
  int d = i % D_; int kv = (i / D_) % KV_; int t = i / (D_ * KV_);
  vt[((size_t)kv * D_ + d) * T_ + t] = (__bf16)v[i];
}

// ---------------- flash attention with block-pooled prob sums ----------------
// grid = (T/64, H); block = 128 (4 waves); wave w handles rows m0 = bx*64 + w*16
__global__ __launch_bounds__(128) void k_attn(const __bf16* __restrict__ qr,
                                              const __bf16* __restrict__ kr,
                                              const __bf16* __restrict__ vt,
                                              float* __restrict__ attn,
                                              float* __restrict__ pooled) {
  __shared__ __bf16 p_lds[4][16][32];
  const int wave = threadIdx.x >> 5;
  const int lane = threadIdx.x & 31;
  const int h = blockIdx.y;
  const int kv = h / G_;
  const int m0 = blockIdx.x * 64 + wave * 16;
  const int mrow = lane & 15;
  const int half = lane >> 4;

  v8f zero = {};
  v8f o[8];
#pragma unroll
  for (int j = 0; j < 8; ++j) o[j] = zero;
  float mrun[8], lrun[8], pbreg[8][2];
#pragma unroll
  for (int i = 0; i < 8; ++i) { mrun[i] = -1e30f; lrun[i] = 0.f; pbreg[i][0] = 0.f; pbreg[i][1] = 0.f; }

  const int tmax = m0 + 15;
  for (int s0 = 0; s0 <= tmax; s0 += 32) {
    // ---- S = Q K^T : two 16x16 tiles, K depth = D_ ----
    v8f sacc0 = zero, sacc1 = zero;
#pragma unroll
    for (int k0 = 0; k0 < D_; k0 += 32) {
      v16bf a, b0, b1;
      const __bf16* arow = qr + ((size_t)(m0 + mrow) * H_ + h) * D_ + k0;
#pragma unroll
      for (int j = 0; j < 8; ++j) a[j]     = arow[half * 16 + j];
#pragma unroll
      for (int j = 0; j < 8; ++j) a[8 + j] = arow[half * 16 + 8 + j];
      const __bf16* b0row = kr + ((size_t)(s0 + mrow) * KV_ + kv) * D_ + k0 + half * 16;
      const __bf16* b1row = kr + ((size_t)(s0 + 16 + mrow) * KV_ + kv) * D_ + k0 + half * 16;
#pragma unroll
      for (int j = 0; j < 16; ++j) { b0[j] = b0row[j]; b1[j] = b1row[j]; }
      sacc0 = __builtin_amdgcn_wmma_f32_16x16x32_bf16(false, a, false, b0, (short)0, sacc0, false, false);
      sacc1 = __builtin_amdgcn_wmma_f32_16x16x32_bf16(false, a, false, b1, (short)0, sacc1, false, false);
    }
    // ---- online softmax update (row = i + 8*half, col = lane%16) ----
    const int blk = s0 >> 6;               // 64-wide block index (NB_=32)
    const int sel = blk >> 4;              // which of the lane's 2 owned slots
    const bool owner = (blk & 15) == mrow; // one owner lane per 16-lane group
#pragma unroll
    for (int i = 0; i < 8; ++i) {
      int t = m0 + i + 8 * half;
      float x0 = sacc0[i] * SCALE_; if (s0 + mrow      > t) x0 = -1e30f;
      float x1 = sacc1[i] * SCALE_; if (s0 + 16 + mrow > t) x1 = -1e30f;
      float m = fmaxf(x0, x1);
      m = fmaxf(m, __shfl_xor(m, 1));
      m = fmaxf(m, __shfl_xor(m, 2));
      m = fmaxf(m, __shfl_xor(m, 4));
      m = fmaxf(m, __shfl_xor(m, 8));
      float mnew = fmaxf(mrun[i], m);
      float f = __expf(mrun[i] - mnew);
      mrun[i] = mnew;
      float e0 = __expf(x0 - mnew);
      float e1 = __expf(x1 - mnew);
      float rs = e0 + e1;
      rs += __shfl_xor(rs, 1); rs += __shfl_xor(rs, 2);
      rs += __shfl_xor(rs, 4); rs += __shfl_xor(rs, 8);
      lrun[i] = lrun[i] * f + rs;
      pbreg[i][0] *= f; pbreg[i][1] *= f;
      if (owner) pbreg[i][sel] += rs;
#pragma unroll
      for (int j = 0; j < 8; ++j) o[j][i] *= f;
      p_lds[wave][i + 8 * half][mrow]      = (__bf16)e0;
      p_lds[wave][i + 8 * half][16 + mrow] = (__bf16)e1;
    }
    __builtin_amdgcn_wave_barrier();
    // ---- re-swizzle P into WMMA A layout ----
    v16bf ap;
#pragma unroll
    for (int j = 0; j < 8; ++j) ap[j]     = p_lds[wave][mrow][half * 16 + j];
#pragma unroll
    for (int j = 0; j < 8; ++j) ap[8 + j] = p_lds[wave][mrow][half * 16 + 8 + j];
    __builtin_amdgcn_wave_barrier();
    // ---- O += P * V  (8 d-tiles of 16) ----
#pragma unroll
    for (int j = 0; j < 8; ++j) {
      v16bf bv;
      const __bf16* vrow = vt + ((size_t)kv * D_ + j * 16 + mrow) * T_ + s0 + half * 16;
#pragma unroll
      for (int e = 0; e < 16; ++e) bv[e] = vrow[e];
      o[j] = __builtin_amdgcn_wmma_f32_16x16x32_bf16(false, ap, false, bv, (short)0, o[j], false, false);
    }
  }
  // ---- finalize: normalize and write attn + pooled ----
#pragma unroll
  for (int i = 0; i < 8; ++i) {
    int t = m0 + i + 8 * half;
    float inv = 1.0f / lrun[i];
#pragma unroll
    for (int j = 0; j < 8; ++j)
      attn[((size_t)t * H_ + h) * D_ + j * 16 + mrow] = o[j][i] * inv;
    pooled[((size_t)h * T_ + t) * NB_ + mrow]      = pbreg[i][0] * inv;
    pooled[((size_t)h * T_ + t) * NB_ + 16 + mrow] = pbreg[i][1] * inv;
  }
}

// ---------------- gate loss: one wave per (kv, t>=START) row ----------------
__global__ void k_gate_loss(const float* __restrict__ pooled, const float* __restrict__ pm,
                            float* __restrict__ loss) {
  int idx = blockIdx.x;
  int kv = idx / (T_ - START_);
  int t = START_ + idx % (T_ - START_);
  int n = threadIdx.x;  // 0..31 == NB_
  float g = fmaxf(pooled[((size_t)(kv * G_ + 0) * T_ + t) * NB_ + n],
                  pooled[((size_t)(kv * G_ + 1) * T_ + t) * NB_ + n]);
  if (t / BS_ < n) g = 0.f;
  float s = g;
  for (int m = 1; m < 32; m <<= 1) s += __shfl_xor(s, m);
  g = g / (s + 1e-9f);
  float x = pm[((size_t)kv * T_ + t) * NB_ + n];
  float mx = x;
  for (int m = 1; m < 32; m <<= 1) mx = fmaxf(mx, __shfl_xor(mx, m));
  float e = __expf(x - mx);
  float se = e;
  for (int m = 1; m < 32; m <<= 1) se += __shfl_xor(se, m);
  float logp = x - mx - __logf(se);
  float kl = (g > 0.f) ? g * (__logf(g) - logp) : 0.f;
  for (int m = 1; m < 32; m <<= 1) kl += __shfl_xor(kl, m);
  if (n == 0) atomicAdd(loss, kl * (1.0f / ((T_ - START_) * NB_)));
}

// ---------------- launch ----------------
extern "C" void kernel_launch(void* const* d_in, const int* in_sizes, int n_in,
                              void* d_out, int out_size, void* d_ws, size_t ws_size,
                              hipStream_t stream) {
  const float* hs      = (const float*)d_in[0];
  const float* Wq      = (const float*)d_in[1];
  const float* Wk      = (const float*)d_in[2];
  const float* Wv      = (const float*)d_in[3];
  const float* Wo      = (const float*)d_in[4];
  const float* qnw     = (const float*)d_in[5];
  const float* knw     = (const float*)d_in[6];
  const float* gate_wq = (const float*)d_in[7];
  const float* gate_wk = (const float*)d_in[8];
  const float* cosT    = (const float*)d_in[9];
  const float* sinT    = (const float*)d_in[10];
  const float* cos_gq  = (const float*)d_in[11];
  const float* sin_gq  = (const float*)d_in[12];
  const float* cos_blk = (const float*)d_in[13];
  const float* sin_blk = (const float*)d_in[14];
  (void)in_sizes; (void)n_in; (void)out_size; (void)ws_size;

  float* out  = (float*)d_out;                 // [T][HID] then loss scalar
  float* loss = out + (size_t)T_ * HID_;

  char* ws = (char*)d_ws;
  size_t off = 0;
  auto alloc = [&](size_t bytes) -> void* {
    void* p = ws + off;
    off += (bytes + 255) & ~(size_t)255;
    return p;
  };
  __bf16* hs_bf  = (__bf16*)alloc((size_t)T_ * HID_ * 2);
  __bf16* Wqt    = (__bf16*)alloc((size_t)HID_ * HID_ * 2);
  __bf16* Wkt    = (__bf16*)alloc((size_t)HID_ * KV_ * D_ * 2);
  __bf16* Wvt    = (__bf16*)alloc((size_t)HID_ * KV_ * D_ * 2);
  __bf16* Wot    = (__bf16*)alloc((size_t)HID_ * HID_ * 2);
  float* qbuf    = (float*)alloc((size_t)T_ * H_ * D_ * 4);
  float* kbuf    = (float*)alloc((size_t)T_ * KV_ * D_ * 4);
  float* vbuf    = (float*)alloc((size_t)T_ * KV_ * D_ * 4);
  float* qpool   = (float*)alloc((size_t)T_ * KV_ * D_ * 4);
  float* kblk    = (float*)alloc((size_t)NB_ * KV_ * D_ * 4);
  float* gq_raw  = (float*)alloc((size_t)T_ * KV_ * GH_ * 4);
  float* gqb     = (float*)alloc((size_t)T_ * KV_ * GH_ * 4);
  float* gk_raw  = (float*)alloc((size_t)NB_ * KV_ * GH_ * 4);
  float* gkb     = (float*)alloc((size_t)NB_ * KV_ * GH_ * 4);
  float* pm      = (float*)alloc((size_t)KV_ * T_ * NB_ * 4);
  __bf16* qr_bf  = (__bf16*)alloc((size_t)T_ * H_ * D_ * 2);
  __bf16* kr_bf  = (__bf16*)alloc((size_t)T_ * KV_ * D_ * 2);
  __bf16* vt_bf  = (__bf16*)alloc((size_t)T_ * KV_ * D_ * 2);
  // reuse dead regions:
  float*  attn_f  = qbuf;             // 16MB, qbuf dead after rope
  __bf16* attn_bf = hs_bf;            // 8MB,  hs_bf dead after projections
  float*  pooled  = (float*)Wqt;      // 4MB,  Wqt dead after Q projection

  const int B = 256;
  // 1) convert + transpose weights
  k_cvt_bf16<<<(T_ * HID_ + B - 1) / B, B, 0, stream>>>(hs, hs_bf, T_ * HID_);
  k_transpose_bf16<<<(HID_ * HID_ + B - 1) / B, B, 0, stream>>>(Wq, Wqt, HID_, H_ * D_);
  k_transpose_bf16<<<(HID_ * KV_ * D_ + B - 1) / B, B, 0, stream>>>(Wk, Wkt, HID_, KV_ * D_);
  k_transpose_bf16<<<(HID_ * KV_ * D_ + B - 1) / B, B, 0, stream>>>(Wv, Wvt, HID_, KV_ * D_);
  k_transpose_bf16<<<(HID_ * HID_ + B - 1) / B, B, 0, stream>>>(Wo, Wot, H_ * D_, HID_);
  // 2) projections (WMMA bf16), 64x64 block tiles
  k_gemm_bf16<<<dim3((H_ * D_) / 64, T_ / 64), 128, 0, stream>>>(hs_bf, Wqt, qbuf, T_, H_ * D_, HID_);
  k_gemm_bf16<<<dim3((KV_ * D_) / 64, T_ / 64), 128, 0, stream>>>(hs_bf, Wkt, kbuf, T_, KV_ * D_, HID_);
  k_gemm_bf16<<<dim3((KV_ * D_) / 64, T_ / 64), 128, 0, stream>>>(hs_bf, Wvt, vbuf, T_, KV_ * D_, HID_);
  // 3) rmsnorm q,k (in place)
  k_rmsnorm<<<T_ * H_, 32, 0, stream>>>(qbuf, qnw, T_ * H_);
  k_rmsnorm<<<T_ * KV_, 32, 0, stream>>>(kbuf, knw, T_ * KV_);
  // 4) pools
  k_qpool<<<(T_ * KV_ * D_ + B - 1) / B, B, 0, stream>>>(qbuf, qpool);
  k_kblk<<<(NB_ * KV_ * D_ + B - 1) / B, B, 0, stream>>>(kbuf, kblk);
  // 5) gate path
  k_small_gemm<<<(T_ * KV_ * GH_ + B - 1) / B, B, 0, stream>>>(qpool, gate_wq, gq_raw, T_ * KV_);
  k_small_gemm<<<(NB_ * KV_ * GH_ + B - 1) / B, B, 0, stream>>>(kblk, gate_wk, gk_raw, NB_ * KV_);
  k_rope_gate<<<(T_ * KV_ * GH_ + B - 1) / B, B, 0, stream>>>(gq_raw, cos_gq, sin_gq, gqb, T_ * KV_);
  k_rope_gate<<<(NB_ * KV_ * GH_ + B - 1) / B, B, 0, stream>>>(gk_raw, cos_blk, sin_blk, gkb, NB_ * KV_);
  k_predict<<<(KV_ * T_ * NB_ + B - 1) / B, B, 0, stream>>>(gqb, gkb, pm);
  // 6) rope q,k -> bf16; v transpose -> bf16
  k_rope_bf16<<<(T_ * H_ * D_ + B - 1) / B, B, 0, stream>>>(qbuf, cosT, sinT, qr_bf, H_);
  k_rope_bf16<<<(T_ * KV_ * D_ + B - 1) / B, B, 0, stream>>>(kbuf, cosT, sinT, kr_bf, KV_);
  k_vt<<<(T_ * KV_ * D_ + B - 1) / B, B, 0, stream>>>(vbuf, vt_bf);
  // 7) flash attention (WMMA) with pooled block sums
  k_attn<<<dim3(T_ / 64, H_), 128, 0, stream>>>(qr_bf, kr_bf, vt_bf, attn_f, pooled);
  // 8) output projection
  k_cvt_bf16<<<(T_ * H_ * D_ + B - 1) / B, B, 0, stream>>>(attn_f, attn_bf, T_ * H_ * D_);
  k_gemm_bf16<<<dim3(HID_ / 64, T_ / 64), 128, 0, stream>>>(attn_bf, Wot, out, T_, HID_, H_ * D_);
  // 9) gate loss
  hipMemsetAsync(loss, 0, sizeof(float), stream);
  k_gate_loss<<<(T_ - START_) * KV_, 32, 0, stream>>>(pooled, pm, loss);
}
